// MultiHeadAttention_37890201486041
// MI455X (gfx1250) — matrix operand order repaired and checked by reading, not compiled, not run on hardware
//
#include <hip/hip_runtime.h>
#include <hip/hip_bf16.h>

typedef __attribute__((ext_vector_type(16))) _Float16 v16h;
typedef __attribute__((ext_vector_type(8)))  _Float16 h8;
typedef __attribute__((ext_vector_type(8)))  float    v8f;
typedef _Float16 half_t;

#define EMB    1024
#define HID    1024
#define NHEADS 16
#define HDIM   64
#define BATCH  2
#define SEQ    2048
#define MROWS  (BATCH*SEQ)   // 4096

union V16 { v16h v; h8 h[2]; };

// ---------------------------------------------------------------------------
// Weight transpose + f32 -> f16:  W[K][N] f32  ->  Wt[N][K] f16   (K=N=1024)
// ---------------------------------------------------------------------------
__global__ void wt_convert_kernel(const float* __restrict__ W, half_t* __restrict__ Wt) {
  __shared__ float tile[32][33];
  const int k0 = blockIdx.y * 32, n0 = blockIdx.x * 32;
  const int tx = threadIdx.x, ty = threadIdx.y;       // block (32, 8)
#pragma unroll
  for (int i = 0; i < 32; i += 8)
    tile[ty + i][tx] = W[(size_t)(k0 + ty + i) * 1024 + n0 + tx];
  __syncthreads();
#pragma unroll
  for (int i = 0; i < 32; i += 8)
    Wt[(size_t)(n0 + ty + i) * 1024 + k0 + tx] = (half_t)tile[tx][ty + i];
}

// ---------------------------------------------------------------------------
// LDS staging helpers: 128 rows x 32 K-cols tile, 256 threads
// ---------------------------------------------------------------------------
__device__ inline void stage_tile(const float* __restrict__ A, int lda, int rowBase,
                                  int k0, half_t (*S)[32], int tid) {
#pragma unroll
  for (int it = 0; it < 4; ++it) {
    const int idx = tid + 256 * it;                 // 0..1023 float4 chunks
    const int row = idx >> 3, c4 = (idx & 7) * 4;
    const float4 f = *(const float4*)&A[(size_t)(rowBase + row) * lda + k0 + c4];
    half_t* d = &S[row][c4];
    d[0] = (half_t)f.x; d[1] = (half_t)f.y; d[2] = (half_t)f.z; d[3] = (half_t)f.w;
  }
}
__device__ inline void stage_tile(const half_t* __restrict__ A, int lda, int rowBase,
                                  int k0, half_t (*S)[32], int tid) {
#pragma unroll
  for (int it = 0; it < 2; ++it) {
    const int idx = tid + 256 * it;                 // 0..511 h8 chunks
    const int row = idx >> 2, c8 = (idx & 3) * 8;
    *(h8*)&S[row][c8] = *(const h8*)&A[(size_t)(rowBase + row) * lda + k0 + c8];
  }
}

// ---------------------------------------------------------------------------
// WMMA GEMM:  C[M,N] = A[M,K] * W[K,N] + bias  (W passed transposed as Bt[N][K] f16)
// MODE 0: Out = float row-major [M,N]
// MODE 1: Out = half head-major [B, NHEADS, SEQ, HDIM], value scaled by `scale`
// MODE 2: Out = half head-major TRANSPOSED [B, NHEADS, HDIM, SEQ] (for V^T)
// Block = 256 threads (8 waves), tile 128x128, BK=32. Wave = 2x4 16x16 tiles.
// ---------------------------------------------------------------------------
template <typename AT, int MODE>
__global__ __launch_bounds__(256)
void gemm_wmma_kernel(const AT* __restrict__ A, const half_t* __restrict__ Bt,
                      const float* __restrict__ bias, float scale,
                      void* __restrict__ Out, int M, int N, int K) {
  __shared__ __align__(32) half_t As[128][32];
  __shared__ __align__(32) half_t Bs[128][32];
  const int tid  = threadIdx.x;
  const int lane = tid & 31, wave = tid >> 5;
  const int wm = wave & 3, wn = wave >> 2;          // 4 x 2 wave grid
  const int lg = lane >> 4, ll = lane & 15;
  const int rowBase = blockIdx.x * 128;
  const int colBase = blockIdx.y * 128;

  const v8f vz = {0, 0, 0, 0, 0, 0, 0, 0};
  v8f acc[2][4];
#pragma unroll
  for (int mi = 0; mi < 2; ++mi)
#pragma unroll
    for (int ni = 0; ni < 4; ++ni) acc[mi][ni] = vz;

  for (int k0 = 0; k0 < K; k0 += 32) {
    __syncthreads();
    stage_tile(A,  K, rowBase, k0, As, tid);
    stage_tile(Bt, K, colBase, k0, Bs, tid);
    __syncthreads();

    v16h af[2], bf[4];
#pragma unroll
    for (int mi = 0; mi < 2; ++mi) {
      const half_t* ap = &As[wm * 32 + mi * 16 + ll][lg * 16];
      V16 t; t.h[0] = *(const h8*)ap; t.h[1] = *(const h8*)(ap + 8);
      af[mi] = t.v;
    }
#pragma unroll
    for (int ni = 0; ni < 4; ++ni)
      bf[ni] = *(const v16h*)&Bs[wn * 64 + ni * 16 + ll][lg * 16];

#pragma unroll
    for (int mi = 0; mi < 2; ++mi)
#pragma unroll
      for (int ni = 0; ni < 4; ++ni)
        acc[mi][ni] = __builtin_amdgcn_wmma_f32_16x16x32_f16(
            false, af[mi], false, bf[ni], (short)0, acc[mi][ni], false, false);
  }

#pragma unroll
  for (int mi = 0; mi < 2; ++mi)
#pragma unroll
    for (int ni = 0; ni < 4; ++ni) {
      const int col = colBase + wn * 64 + ni * 16 + ll;
      const float bcol = bias[col];
#pragma unroll
      for (int r = 0; r < 8; ++r) {
        const int row = rowBase + wm * 32 + mi * 16 + r + 8 * lg;
        const float val = acc[mi][ni][r] + bcol;
        const int bb = row >> 11, s = row & (SEQ - 1);      // row = b*SEQ + s
        const int h  = col >> 6,  d = col & (HDIM - 1);     // col = h*64 + d
        if (MODE == 0) {
          ((float*)Out)[(size_t)row * N + col] = val;
        } else if (MODE == 1) {
          ((half_t*)Out)[(((size_t)(bb * NHEADS + h)) * SEQ + s) * HDIM + d] =
              (half_t)(val * scale);
        } else {
          ((half_t*)Out)[(((size_t)(bb * NHEADS + h)) * HDIM + d) * SEQ + s] =
              (half_t)(val * scale);
        }
      }
    }
}

// ---------------------------------------------------------------------------
// Flash attention, transposed formulation.
//   S^T = K * Q^T  (A-frag = K rows from global, B-frag = Q^T resident)
//   O^T = V^T * P^T (A-frag = V^T rows from global, B-frag = P^T via LDS)
// C-layout of S^T: lane = query column, in-lane VGPR axis = keys -> softmax
// reductions are in-register + ONE __shfl_xor(,16) each for max and sum;
// m / l / alpha are one scalar per lane.
// One block per (q-tile of 64, head, batch); 4 waves, wave = 16 queries.
// Q pre-scaled by 1/sqrt(64) at projection time.
// ---------------------------------------------------------------------------
__global__ __launch_bounds__(128)
void attn_kernel(const half_t* __restrict__ Q, const half_t* __restrict__ K,
                 const half_t* __restrict__ Vt, half_t* __restrict__ Ctx) {
  __shared__ __align__(32) half_t Pq[4][16][80];   // per-wave P^T [query][key], padded
  const int tid  = threadIdx.x;
  const int lane = tid & 31, wave = tid >> 5;
  const int lg = lane >> 4, ll = lane & 15;
  const int qt = blockIdx.x, h = blockIdx.y, b = blockIdx.z;
  const size_t headOff = ((size_t)(b * NHEADS + h)) * SEQ * HDIM; // same for [S,D] and [D,S]
  const half_t* Qh = Q  + headOff;   // [S][64]
  const half_t* Kh = K  + headOff;   // [S][64]
  const half_t* Vh = Vt + headOff;   // [64][S]  (transposed)
  const int query = qt * 64 + wave * 16 + ll;

  // Q^T B-fragments (col n = query, k = d), resident for whole kernel
  v16h bq[2];
  bq[0] = *(const v16h*)&Qh[(size_t)query * HDIM + lg * 16];
  bq[1] = *(const v16h*)&Qh[(size_t)query * HDIM + 32 + lg * 16];

  const v8f vz = {0, 0, 0, 0, 0, 0, 0, 0};
  float m = -__builtin_inff(), l = 0.f;
  v8f o[4];                                        // O^T frags: rows d, col = query
#pragma unroll
  for (int dt = 0; dt < 4; ++dt) o[dt] = vz;

  for (int kb = 0; kb < SEQ; kb += 64) {
    if (kb + 64 < SEQ) {                           // prefetch next K / V^T tiles
      __builtin_prefetch(&Kh[(size_t)(kb + 64 + (tid & 63)) * HDIM], 0, 1);
      __builtin_prefetch(&Vh[(size_t)(tid & 63) * SEQ + kb + 64], 0, 1);
    }

    // --- scores S^T: 4 key sub-tiles of 16, K A-frags straight from global ---
    v8f s[4];
#pragma unroll
    for (int kt = 0; kt < 4; ++kt) {
      const half_t* kp = &Kh[(size_t)(kb + kt * 16 + ll) * HDIM];
      V16 a0, a1;
      a0.h[0] = *(const h8*)&kp[lg * 16];      a0.h[1] = *(const h8*)&kp[lg * 16 + 8];
      a1.h[0] = *(const h8*)&kp[32 + lg * 16]; a1.h[1] = *(const h8*)&kp[32 + lg * 16 + 8];
      v8f c = vz;
      c = __builtin_amdgcn_wmma_f32_16x16x32_f16(false, a0.v, false, bq[0], (short)0, c, false, false);
      c = __builtin_amdgcn_wmma_f32_16x16x32_f16(false, a1.v, false, bq[1], (short)0, c, false, false);
      s[kt] = c;
    }

    // --- online softmax: in-register reduce + one xor-16 shuffle each ---
    float mx = s[0][0];
#pragma unroll
    for (int kt = 0; kt < 4; ++kt)
#pragma unroll
      for (int r = 0; r < 8; ++r) mx = fmaxf(mx, s[kt][r]);
    mx = fmaxf(mx, __shfl_xor(mx, 16));
    const float mnew  = fmaxf(m, mx);
    const float alpha = __expf(m - mnew);
    m = mnew;

    float rs = 0.f;
#pragma unroll
    for (int kt = 0; kt < 4; ++kt)
#pragma unroll
      for (int r = 0; r < 8; ++r) {
        const float p = __expf(s[kt][r] - mnew);
        s[kt][r] = p;
        rs += p;
      }
    rs += __shfl_xor(rs, 16);
    l = l * alpha + rs;
#pragma unroll
    for (int dt = 0; dt < 4; ++dt)
#pragma unroll
      for (int r = 0; r < 8; ++r) o[dt][r] = o[dt][r] * alpha;

    // --- P^T to wave-local LDS: per lane 8 consecutive keys -> packed b128 ---
#pragma unroll
    for (int kt = 0; kt < 4; ++kt) {
      h8 ph;
#pragma unroll
      for (int r = 0; r < 8; ++r) ph[r] = (half_t)s[kt][r];
      *(h8*)&Pq[wave][ll][kt * 16 + 8 * lg] = ph;
    }
    __syncthreads();
    const v16h bp0 = *(const v16h*)&Pq[wave][ll][lg * 16];        // keys 0..31
    const v16h bp1 = *(const v16h*)&Pq[wave][ll][32 + lg * 16];   // keys 32..63

    // --- O^T += V^T * P^T, V^T A-frags straight from global ---
#pragma unroll
    for (int dt = 0; dt < 4; ++dt) {
      const half_t* vp = &Vh[(size_t)(dt * 16 + ll) * SEQ + kb];
      V16 a0, a1;
      a0.h[0] = *(const h8*)&vp[lg * 16];      a0.h[1] = *(const h8*)&vp[lg * 16 + 8];
      a1.h[0] = *(const h8*)&vp[32 + lg * 16]; a1.h[1] = *(const h8*)&vp[32 + lg * 16 + 8];
      o[dt] = __builtin_amdgcn_wmma_f32_16x16x32_f16(false, a0.v, false, bp0, (short)0, o[dt], false, false);
      o[dt] = __builtin_amdgcn_wmma_f32_16x16x32_f16(false, a1.v, false, bp1, (short)0, o[dt], false, false);
    }
  }

  // --- normalize, write context: lane owns one Ctx row -> packed b128 stores ---
  const float inv = 1.0f / l;
  const size_t rowoff = ((size_t)(b * SEQ + query)) * HID + h * HDIM;
#pragma unroll
  for (int dt = 0; dt < 4; ++dt) {
    h8 oc;
#pragma unroll
    for (int r = 0; r < 8; ++r) oc[r] = (half_t)(o[dt][r] * inv);
    *(h8*)&Ctx[rowoff + dt * 16 + 8 * lg] = oc;    // d = dt*16 + 8*lg + r
  }
}

// ---------------------------------------------------------------------------
extern "C" void kernel_launch(void* const* d_in, const int* in_sizes, int n_in,
                              void* d_out, int out_size, void* d_ws, size_t ws_size,
                              hipStream_t stream) {
  const float* q  = (const float*)d_in[0];
  const float* k  = (const float*)d_in[1];
  const float* v  = (const float*)d_in[2];
  const float* Wq = (const float*)d_in[3];
  const float* bq = (const float*)d_in[4];
  const float* Wk = (const float*)d_in[5];
  const float* bk = (const float*)d_in[6];
  const float* Wv = (const float*)d_in[7];
  const float* bv = (const float*)d_in[8];
  const float* Wo = (const float*)d_in[9];
  const float* bo = (const float*)d_in[10];
  float* out = (float*)d_out;

  // Workspace layout (40 MB total):
  char* ws = (char*)d_ws;
  const size_t WSZ = (size_t)1024 * 1024 * sizeof(half_t);        // 2 MB each
  half_t* Wq_t = (half_t*)(ws + 0 * WSZ);
  half_t* Wk_t = (half_t*)(ws + 1 * WSZ);
  half_t* Wv_t = (half_t*)(ws + 2 * WSZ);
  half_t* Wo_t = (half_t*)(ws + 3 * WSZ);
  const size_t QSZ = (size_t)MROWS * HID * sizeof(half_t);        // 8 MB each
  half_t* Qh  = (half_t*)(ws + 4 * WSZ);                          // [B,H,S,64]
  half_t* Kh  = (half_t*)(ws + 4 * WSZ + 1 * QSZ);                // [B,H,S,64]
  half_t* Vh  = (half_t*)(ws + 4 * WSZ + 2 * QSZ);                // [B,H,64,S] (transposed)
  half_t* Ctx = (half_t*)(ws + 4 * WSZ + 3 * QSZ);                // [B*S, HID]

  // 1) Weight transpose + convert
  dim3 wtb(32, 8), wtg(32, 32);
  wt_convert_kernel<<<wtg, wtb, 0, stream>>>(Wq, Wq_t);
  wt_convert_kernel<<<wtg, wtb, 0, stream>>>(Wk, Wk_t);
  wt_convert_kernel<<<wtg, wtb, 0, stream>>>(Wv, Wv_t);
  wt_convert_kernel<<<wtg, wtb, 0, stream>>>(Wo, Wo_t);

  // 2) Q/K/V projections (WMMA); Q pre-scaled by 1/sqrt(64); V written transposed
  dim3 gg(MROWS / 128, HID / 128);                                // 32 x 8
  gemm_wmma_kernel<float, 1><<<gg, 256, 0, stream>>>(q, Wq_t, bq, 0.125f, Qh, MROWS, HID, EMB);
  gemm_wmma_kernel<float, 1><<<gg, 256, 0, stream>>>(k, Wk_t, bk, 1.0f,   Kh, MROWS, HID, EMB);
  gemm_wmma_kernel<float, 2><<<gg, 256, 0, stream>>>(v, Wv_t, bv, 1.0f,   Vh, MROWS, HID, EMB);

  // 3) Flash attention (WMMA scores + WMMA PV, transposed online softmax)
  dim3 ag(SEQ / 64, NHEADS, BATCH);                               // 32 x 16 x 2
  attn_kernel<<<ag, 128, 0, stream>>>(Qh, Kh, Vh, Ctx);

  // 4) Output projection (WMMA) + bias -> f32 out
  gemm_wmma_kernel<half_t, 0><<<gg, 256, 0, stream>>>(Ctx, Wo_t, bo, 1.0f, out, MROWS, EMB, HID);
}
